// DescriptorNetwork_9174050144805
// MI455X (gfx1250) — matrix operand order repaired and checked by reading, not compiled, not run on hardware
//
#include <hip/hip_runtime.h>
#include <stdint.h>

typedef __attribute__((ext_vector_type(16))) __bf16 v16bf;
typedef __attribute__((ext_vector_type(8)))  float  v8f;

#define FEA 64
#define HID 256
#define NGRAPH 3
#define NHEADS 3

// ---------- helpers ----------
__device__ inline uint16_t bf_bits(float f) {
  union { float f; uint32_t u; } a; a.f = f;
  uint32_t u = a.u;
  u += 0x7fffu + ((u >> 16) & 1u);   // round-to-nearest-even
  return (uint16_t)(u >> 16);
}
__device__ inline __bf16 f2bf(float f) {
  union { uint16_t s; __bf16 b; } r; r.s = bf_bits(f); return r.b;
}
// monotone float<->uint encoding for atomicMax-based segment max
__device__ inline unsigned encf(float f) {
  unsigned u = __float_as_uint(f);
  return (u & 0x80000000u) ? ~u : (u | 0x80000000u);
}
__device__ inline float decf(unsigned u) {
  u = (u & 0x80000000u) ? (u & 0x7fffffffu) : ~u;
  return __uint_as_float(u);
}

union V16BF { v16bf v; uint32_t d[8]; };

__device__ inline v8f vzero() {
  v8f z;
#pragma unroll
  for (int i = 0; i < 8; ++i) z[i] = 0.f;
  return z;
}

// Load WMMA A operand (16xK bf16, row-major in LDS, stride ldaDw dwords).
// Layout per ISA: lane<16 holds row M=lane, K in {k0..k0+7} U {k0+16..k0+23};
// lane>=16 holds row M=lane-16, K in {k0+8..k0+15} U {k0+24..k0+31}.
__device__ inline v16bf lds_load_A(const uint32_t* base, int ldaDw, int k0) {
  int lane = threadIdx.x & 31;
  const uint32_t* row = base + (lane & 15) * ldaDw;
  int kb = (k0 >> 1) + ((lane & 16) ? 4 : 0);
  V16BF r;
#pragma unroll
  for (int d = 0; d < 4; ++d) r.d[d] = row[kb + d];
#pragma unroll
  for (int d = 0; d < 4; ++d) r.d[4 + d] = row[kb + 8 + d];
  return r.v;
}

// Load pre-swizzled WMMA B operand tile (256 dwords, lane*8 contiguous).
__device__ inline v16bf load_B(const uint32_t* __restrict__ tile) {
  int lane = threadIdx.x & 31;
  const uint32_t* p = tile + lane * 8;
  V16BF r;
#pragma unroll
  for (int d = 0; d < 8; ++d) r.d[d] = p[d];
  return r.v;
}

__device__ inline v8f wmma_bf16(v16bf a, v16bf b, v8f c) {
  return __builtin_amdgcn_wmma_f32_16x16x32_bf16(false, a, false, b, (short)0, c,
                                                 false, false);
}

// ---------- utility kernels ----------
__global__ void k_zero(uint32_t* __restrict__ p, long n) {
  long i = blockIdx.x * (long)blockDim.x + threadIdx.x;
  if (i < n) p[i] = 0u;
}

__global__ void k_scale(float* __restrict__ o, long n) {
  long i = blockIdx.x * (long)blockDim.x + threadIdx.x;
  if (i < n) o[i] *= (1.f / 3.f);
}

__global__ void k_residual(float* __restrict__ x, __bf16* __restrict__ xb,
                           float* __restrict__ acc, long n) {
  long i = blockIdx.x * (long)blockDim.x + threadIdx.x;
  if (i >= n) return;
  float v = x[i] + acc[i] * (1.f / 3.f);
  x[i] = v;
  xb[i] = f2bf(v);
  acc[i] = 0.f;   // ready for next layer's heads
}

// Pack f32 (K,N) row-major weight into bf16 WMMA-B tile layout.
// tile = nt*KS + ks ; within tile: dword = lane*8 + d
__global__ void k_pack(const float* __restrict__ src, int KS, int Ncols,
                       uint32_t* __restrict__ dst, int total) {
  int g = blockIdx.x * 256 + threadIdx.x;
  if (g >= total) return;
  int tile = g >> 8;
  int within = g & 255;
  int lane = within >> 3;
  int d = within & 7;
  int nt = tile / KS;
  int ks = tile - nt * KS;
  int n = nt * 16 + (lane & 15);
  int k = ks * 32 + ((lane & 16) ? 8 : 0) + ((d < 4) ? 2 * d : 8 + 2 * d);
  uint32_t lo = bf_bits(src[(size_t)k * Ncols + n]);
  uint32_t hi = bf_bits(src[(size_t)(k + 1) * Ncols + n]);
  dst[g] = lo | (hi << 16);
}

// Embedding: x[:, :63] = elem_fea @ emb_w + emb_b ; x[:,63] = elem_weights
__global__ __launch_bounds__(64) void k_embed(
    const float* __restrict__ elem_fea, const float* __restrict__ elem_w,
    const float* __restrict__ emb_w, const float* __restrict__ emb_b,
    float* __restrict__ x, __bf16* __restrict__ xb, int N, int EMBD) {
  __shared__ float row[256];
  int n = blockIdx.x;
  int t = threadIdx.x;
  for (int j = t; j < EMBD; j += 64) row[j] = elem_fea[(size_t)n * EMBD + j];
  __syncthreads();
  float s;
  if (t < FEA - 1) {
    s = emb_b[t];
#pragma unroll 4
    for (int j = 0; j < EMBD; ++j) s += row[j] * emb_w[j * (FEA - 1) + t];
  } else {
    s = elem_w[n];
  }
  x[(size_t)n * FEA + t] = s;
  xb[(size_t)n * FEA + t] = f2bf(s);
}

// ---------- gather into LDS (bf16) ----------
template <bool EDGE, int LDA>
__device__ inline void gather_fea(__bf16* feaLDS, const __bf16* __restrict__ xb,
                                  const int* __restrict__ sidx,
                                  const int* __restrict__ nidx, int e0, int M) {
  int t = threadIdx.x;
  int m = t >> 3, ch = t & 7;
  int e = e0 + m;
  if (e > M - 1) e = M - 1;
  if constexpr (EDGE) {
    int col = ch * 16;                    // 16 bf16 per thread, 2*FEA cols
    int node = (col < FEA) ? sidx[e] : nidx[e];
    int c0 = col & (FEA - 1);
    const uint4* src = (const uint4*)(xb + (size_t)node * FEA + c0);
    uint4* dst = (uint4*)(feaLDS + m * LDA + col);
    dst[0] = src[0];
    dst[1] = src[1];
  } else {
    int col = ch * 8;                     // 8 bf16 per thread, FEA cols
    const uint4* src = (const uint4*)(xb + (size_t)e * FEA + col);
    uint4* dst = (uint4*)(feaLDS + m * LDA + col);
    dst[0] = src[0];
  }
}

// ---------- gate kernel: GEMM1 (WMMA) + leakyReLU + 256->1 dot + seg-max ----------
template <int KSTEPS, bool EDGE>
__global__ __launch_bounds__(128) void k_gate(
    const __bf16* __restrict__ xb, const int* __restrict__ sidx,
    const int* __restrict__ nidx, const uint32_t* __restrict__ w1p,
    const float* __restrict__ b1, const float* __restrict__ w2,
    const float* __restrict__ b2, float* __restrict__ gate_e,
    unsigned* __restrict__ gmax, int M) {
  constexpr int LDA = EDGE ? 136 : 72;   // padded bf16 stride (16B multiple)
  __shared__ __align__(16) __bf16 feaLDS[16 * 136];
  __shared__ float hiddenF[16 * 260];
  const int t = threadIdx.x;
  const int e0 = blockIdx.x * 16;

  gather_fea<EDGE, LDA>(feaLDS, xb, sidx, nidx, e0, M);
  __syncthreads();

  const int wave = t >> 5, lane = t & 31;
  v8f acc[4];
#pragma unroll
  for (int j = 0; j < 4; ++j) acc[j] = vzero();

#pragma unroll
  for (int ks = 0; ks < KSTEPS; ++ks) {
    v16bf a = lds_load_A((const uint32_t*)feaLDS, LDA / 2, ks * 32);
#pragma unroll
    for (int j = 0; j < 4; ++j) {
      v16bf b = load_B(w1p + (size_t)((wave * 4 + j) * KSTEPS + ks) * 256);
      acc[j] = wmma_bf16(a, b, acc[j]);
    }
  }

  const int mbase = (lane & 16) ? 8 : 0;
#pragma unroll
  for (int j = 0; j < 4; ++j) {
    int n = (wave * 4 + j) * 16 + (lane & 15);
    float bias = b1[n];
#pragma unroll
    for (int r = 0; r < 8; ++r) {
      float v = acc[j][r] + bias;
      v = v > 0.f ? v : 0.01f * v;       // LeakyReLU(0.01)
      hiddenF[(mbase + r) * 260 + n] = v;
    }
  }
  __syncthreads();

  if (t < 16) {
    int e = e0 + t;
    if (e < M) {
      float g = b2[0];
      const float* h = hiddenF + t * 260;
#pragma unroll 8
      for (int k = 0; k < HID; ++k) g += h[k] * w2[k];
      gate_e[e] = g;
      atomicMax(&gmax[sidx[e]], encf(g));
    }
  }
}

// ---------- denominator: exp(g-gmax)*w^p, segment sum ----------
template <bool EDGE>
__global__ void k_den(float* __restrict__ gate_e, const int* __restrict__ sidx,
                      const int* __restrict__ nidx,
                      const float* __restrict__ elemw,
                      const unsigned* __restrict__ gmax, float* __restrict__ den,
                      const float* __restrict__ powp, int M) {
  int e = blockIdx.x * 256 + threadIdx.x;
  if (e >= M) return;
  int s = sidx[e];
  float mx = decf(gmax[s]);
  float w = EDGE ? elemw[nidx[e]] : elemw[e];
  float g = __expf(gate_e[e] - mx) * __powf(w, powp[0]);
  gate_e[e] = g;
  atomicAdd(&den[s], g);
}

// ---------- msg kernel: GEMM1 (WMMA) + GEMM2 (WMMA) + gated scatter-add ----------
template <int KSTEPS, bool EDGE>
__global__ __launch_bounds__(128) void k_msg(
    const __bf16* __restrict__ xb, const int* __restrict__ sidx,
    const int* __restrict__ nidx, const uint32_t* __restrict__ w1p,
    const float* __restrict__ b1, const uint32_t* __restrict__ w2p,
    const float* __restrict__ b2, const float* __restrict__ gate_e,
    const float* __restrict__ den, float* __restrict__ out, int M) {
  constexpr int LDA = EDGE ? 136 : 72;
  __shared__ __align__(16) __bf16 feaLDS[16 * 136];
  __shared__ __align__(16) __bf16 hiddenB[16 * 264];
  __shared__ float coefS[16];
  __shared__ int idxS[16];
  const int t = threadIdx.x;
  const int e0 = blockIdx.x * 16;

  gather_fea<EDGE, LDA>(feaLDS, xb, sidx, nidx, e0, M);
  __syncthreads();

  const int wave = t >> 5, lane = t & 31;
  v8f acc[4];
#pragma unroll
  for (int j = 0; j < 4; ++j) acc[j] = vzero();

#pragma unroll
  for (int ks = 0; ks < KSTEPS; ++ks) {
    v16bf a = lds_load_A((const uint32_t*)feaLDS, LDA / 2, ks * 32);
#pragma unroll
    for (int j = 0; j < 4; ++j) {
      v16bf b = load_B(w1p + (size_t)((wave * 4 + j) * KSTEPS + ks) * 256);
      acc[j] = wmma_bf16(a, b, acc[j]);
    }
  }

  const int mbase = (lane & 16) ? 8 : 0;
#pragma unroll
  for (int j = 0; j < 4; ++j) {
    int n = (wave * 4 + j) * 16 + (lane & 15);
    float bias = b1[n];
#pragma unroll
    for (int r = 0; r < 8; ++r) {
      float v = acc[j][r] + bias;
      v = v > 0.f ? v : 0.01f * v;
      hiddenB[(mbase + r) * 264 + n] = f2bf(v);
    }
  }
  if (t < 16) {
    int e = e0 + t;
    if (e < M) {
      int s = sidx[e];
      idxS[t] = s;
      coefS[t] = gate_e[e] / (den[s] + 1e-10f);
    } else {
      idxS[t] = 0;
      coefS[t] = 0.f;
    }
  }
  __syncthreads();

  // GEMM2: hidden (16x256) @ w2 (256x64); each wave owns 16 output cols
  v8f accM = vzero();
#pragma unroll
  for (int ks = 0; ks < 8; ++ks) {
    v16bf a = lds_load_A((const uint32_t*)hiddenB, 132, ks * 32);
    v16bf b = load_B(w2p + (size_t)(wave * 8 + ks) * 256);
    accM = wmma_bf16(a, b, accM);
  }
  int n = wave * 16 + (lane & 15);
  float bias = b2[n];
#pragma unroll
  for (int r = 0; r < 8; ++r) {
    int m = mbase + r;
    if (e0 + m < M) {
      float v = (accM[r] + bias) * coefS[m];
      atomicAdd(&out[(size_t)idxS[m] * FEA + n], v);
    }
  }
}

// ---------- host side ----------
extern "C" void kernel_launch(void* const* d_in, const int* in_sizes, int n_in,
                              void* d_out, int out_size, void* d_ws,
                              size_t ws_size, hipStream_t stream) {
  (void)n_in; (void)ws_size;
  const float* elem_weights = (const float*)d_in[0];
  const float* elem_fea     = (const float*)d_in[1];
  const int*   self_idx     = (const int*)d_in[2];
  const int*   nbr_idx      = (const int*)d_in[3];
  const int*   cry_idx      = (const int*)d_in[4];
  const float* emb_w        = (const float*)d_in[5];
  const float* emb_b        = (const float*)d_in[6];
  const float* g_gate_w1    = (const float*)d_in[7];
  const float* g_gate_b1    = (const float*)d_in[8];
  const float* g_gate_w2    = (const float*)d_in[9];
  const float* g_gate_b2    = (const float*)d_in[10];
  const float* g_msg_w1     = (const float*)d_in[11];
  const float* g_msg_b1     = (const float*)d_in[12];
  const float* g_msg_w2     = (const float*)d_in[13];
  const float* g_msg_b2     = (const float*)d_in[14];
  const float* g_pow        = (const float*)d_in[15];
  const float* c_gate_w1    = (const float*)d_in[16];
  const float* c_gate_b1    = (const float*)d_in[17];
  const float* c_gate_w2    = (const float*)d_in[18];
  const float* c_gate_b2    = (const float*)d_in[19];
  const float* c_msg_w1     = (const float*)d_in[20];
  const float* c_msg_b1     = (const float*)d_in[21];
  const float* c_msg_w2     = (const float*)d_in[22];
  const float* c_msg_b2     = (const float*)d_in[23];
  const float* c_pow        = (const float*)d_in[24];

  const int N = in_sizes[0];          // nodes
  const int E = in_sizes[2];          // edges
  const int C = out_size / FEA;       // crystals
  const int EMBD = in_sizes[1] / N;   // 200

  // workspace carve-up
  char* p = (char*)d_ws;
  auto carve = [&](size_t bytes) -> char* {
    char* r = p;
    p += (bytes + 255) & ~(size_t)255;
    return r;
  };
  float*    x      = (float*)carve((size_t)N * FEA * 4);
  __bf16*   xb     = (__bf16*)carve((size_t)N * FEA * 2);
  float*    acc    = (float*)carve((size_t)N * FEA * 4);
  float*    gate_e = (float*)carve((size_t)E * 4);
  unsigned* gmax   = (unsigned*)carve((size_t)N * 4);
  float*    den    = (float*)carve((size_t)N * 4);
  uint32_t* pw_ggate = (uint32_t*)carve((size_t)9 * 64 * 256 * 4);
  uint32_t* pw_gmsg1 = (uint32_t*)carve((size_t)9 * 64 * 256 * 4);
  uint32_t* pw_gmsg2 = (uint32_t*)carve((size_t)9 * 32 * 256 * 4);
  uint32_t* pw_cgate = (uint32_t*)carve((size_t)3 * 32 * 256 * 4);
  uint32_t* pw_cmsg1 = (uint32_t*)carve((size_t)3 * 32 * 256 * 4);
  uint32_t* pw_cmsg2 = (uint32_t*)carve((size_t)3 * 32 * 256 * 4);

  // embedding
  k_embed<<<N, 64, 0, stream>>>(elem_fea, elem_weights, emb_w, emb_b, x, xb, N,
                                EMBD);

  // pack weights into WMMA-B bf16 layout
  auto pack = [&](const float* src, int K, int Ncols, uint32_t* dst) {
    int KS = K / 32;
    int tot = KS * (Ncols / 16) * 256;
    k_pack<<<(tot + 255) / 256, 256, 0, stream>>>(src, KS, Ncols, dst, tot);
  };
  for (int l = 0; l < NGRAPH; ++l)
    for (int h = 0; h < NHEADS; ++h) {
      int i = l * NHEADS + h;
      pack(g_gate_w1 + (size_t)i * 2 * FEA * HID, 2 * FEA, HID,
           pw_ggate + (size_t)i * 64 * 256);
      pack(g_msg_w1 + (size_t)i * 2 * FEA * HID, 2 * FEA, HID,
           pw_gmsg1 + (size_t)i * 64 * 256);
      pack(g_msg_w2 + (size_t)i * HID * FEA, HID, FEA,
           pw_gmsg2 + (size_t)i * 32 * 256);
    }
  for (int h = 0; h < NHEADS; ++h) {
    pack(c_gate_w1 + (size_t)h * FEA * HID, FEA, HID,
         pw_cgate + (size_t)h * 32 * 256);
    pack(c_msg_w1 + (size_t)h * FEA * HID, FEA, HID,
         pw_cmsg1 + (size_t)h * 32 * 256);
    pack(c_msg_w2 + (size_t)h * HID * FEA, HID, FEA,
         pw_cmsg2 + (size_t)h * 32 * 256);
  }

  auto zero = [&](void* ptr, long n) {
    k_zero<<<(int)((n + 255) / 256), 256, 0, stream>>>((uint32_t*)ptr, n);
  };
  zero(acc, (long)N * FEA);

  const int EB = (E + 15) / 16;
  const int NB = (N + 15) / 16;

  // graph layers
  for (int l = 0; l < NGRAPH; ++l) {
    for (int h = 0; h < NHEADS; ++h) {
      int i = l * NHEADS + h;
      zero(gmax, N);
      zero(den, N);
      k_gate<4, true><<<EB, 128, 0, stream>>>(
          xb, self_idx, nbr_idx, pw_ggate + (size_t)i * 64 * 256,
          g_gate_b1 + (size_t)i * HID, g_gate_w2 + (size_t)i * HID,
          g_gate_b2 + i, gate_e, gmax, E);
      k_den<true><<<(E + 255) / 256, 256, 0, stream>>>(
          gate_e, self_idx, nbr_idx, elem_weights, gmax, den, g_pow + i, E);
      k_msg<4, true><<<EB, 128, 0, stream>>>(
          xb, self_idx, nbr_idx, pw_gmsg1 + (size_t)i * 64 * 256,
          g_msg_b1 + (size_t)i * HID, pw_gmsg2 + (size_t)i * 32 * 256,
          g_msg_b2 + (size_t)i * FEA, gate_e, den, acc, E);
    }
    k_residual<<<(int)(((long)N * FEA + 255) / 256), 256, 0, stream>>>(
        x, xb, acc, (long)N * FEA);
  }

  // crystal pooling
  zero(d_out, (long)C * FEA);
  for (int h = 0; h < NHEADS; ++h) {
    zero(gmax, C);
    zero(den, C);
    k_gate<2, false><<<NB, 128, 0, stream>>>(
        xb, cry_idx, nullptr, pw_cgate + (size_t)h * 32 * 256,
        c_gate_b1 + (size_t)h * HID, c_gate_w2 + (size_t)h * HID, c_gate_b2 + h,
        gate_e, gmax, N);
    k_den<false><<<(N + 255) / 256, 256, 0, stream>>>(
        gate_e, cry_idx, nullptr, elem_weights, gmax, den, c_pow + h, N);
    k_msg<2, false><<<NB, 128, 0, stream>>>(
        xb, cry_idx, nullptr, pw_cmsg1 + (size_t)h * 32 * 256,
        c_msg_b1 + (size_t)h * HID, pw_cmsg2 + (size_t)h * 32 * 256,
        c_msg_b2 + (size_t)h * FEA, gate_e, den, (float*)d_out, N);
  }
  k_scale<<<(int)(((long)C * FEA + 255) / 256), 256, 0, stream>>>(
      (float*)d_out, (long)C * FEA);
}
